// LSTMTagger_26328149525251
// MI455X (gfx1250) — matrix operand order repaired and checked by reading, not compiled, not run on hardware
//
#include <hip/hip_runtime.h>
#include <hip/hip_bf16.h>
#include <stdint.h>
#include <stddef.h>

// ---------------- problem constants (match reference) ----------------
#define S_LEN   4096
#define L_CH    16
#define CE      30
#define CHH     25
#define WE      300
#define DIN_PAD 352     // 350 padded to multiple of 32
#define HID     512
#define G4      2048    // 4*HID
#define D2H     1024    // 2*HID
#define NTAG    52
#define NT_PAD  64

// ---------------- WMMA types ----------------
typedef __attribute__((ext_vector_type(16))) __bf16 v16bf;
typedef __attribute__((ext_vector_type(8)))  float  v8f;

union V16U { v16bf v; uint4 q[2]; };

__device__ __forceinline__ unsigned short f2bf(float f) {
  unsigned int u = __float_as_uint(f);
  u += 0x7FFFu + ((u >> 16) & 1u);        // round-to-nearest-even
  return (unsigned short)(u >> 16);
}
__device__ __forceinline__ float sigmoidf_(float x) {
  return 1.0f / (1.0f + __expf(-x));
}

// =====================================================================
// K1: gather word embeddings -> x_bf[s][0..299], zero pad cols 350..351
// =====================================================================
__global__ void __launch_bounds__(128) gather_x_kernel(
    const int* __restrict__ words, const float* __restrict__ word_emb,
    unsigned short* __restrict__ x_bf)
{
  const int s = blockIdx.x;
  const int w = words[s];
  const float* src = word_emb + (size_t)w * WE;
  unsigned short* dst = x_bf + (size_t)s * DIN_PAD;
  for (int k = threadIdx.x; k < WE; k += 128) dst[k] = f2bf(src[k]);
  if (threadIdx.x < 2) dst[350 + threadIdx.x] = 0;
}

// =====================================================================
// K2: fp32 -> bf16 with zero padding (for Wih_f/b and W_tag)
// =====================================================================
__global__ void __launch_bounds__(256) convert_pad_kernel(
    const float* __restrict__ src, int srows, int scols,
    unsigned short* __restrict__ dst, int drows, int dcols)
{
  int idx = blockIdx.x * 256 + threadIdx.x;
  if (idx >= drows * dcols) return;
  int r = idx / dcols, c = idx - r * dcols;
  dst[idx] = (r < srows && c < scols) ? f2bf(src[r * scols + c]) : (unsigned short)0;
}

// =====================================================================
// K3: per-word char BiLSTM. Block = 256 thr = 8 waves = 4 words x 2 dirs.
// Wave lane j (<25) owns hidden index j fully (rows j, 25+j, 50+j, 75+j).
// Writes final h into x_bf[word][300 + dir*25 + j] as bf16.
// =====================================================================
__global__ void __launch_bounds__(256) char_bilstm_kernel(
    const int* __restrict__ chars, const int* __restrict__ char_lens,
    const float* __restrict__ char_emb,
    const float* __restrict__ cWih_f, const float* __restrict__ cWhh_f, const float* __restrict__ cb_f,
    const float* __restrict__ cWih_b, const float* __restrict__ cWhh_b, const float* __restrict__ cb_b,
    unsigned short* __restrict__ x_bf)
{
  __shared__ float sWih[2][100 * CE];
  __shared__ float sWhh[2][100 * CHH];
  __shared__ float sb[2][100];
  __shared__ float se[4][L_CH * CE];
  __shared__ float sh[4][2][32];

  const int tid = threadIdx.x;
  for (int i = tid; i < 100 * CE; i += 256) { sWih[0][i] = cWih_f[i]; sWih[1][i] = cWih_b[i]; }
  for (int i = tid; i < 100 * CHH; i += 256) { sWhh[0][i] = cWhh_f[i]; sWhh[1][i] = cWhh_b[i]; }
  if (tid < 100) { sb[0][tid] = cb_f[tid]; sb[1][tid] = cb_b[tid]; }

  const int word0 = blockIdx.x * 4;
  for (int i = tid; i < 4 * L_CH * CE; i += 256) {
    int wl = i / (L_CH * CE); int rem = i - wl * (L_CH * CE);
    int t = rem / CE; int k = rem - t * CE;
    int ch = chars[(word0 + wl) * L_CH + t];
    se[wl][t * CE + k] = char_emb[ch * CE + k];
  }
  for (int i = tid; i < 4 * 2 * 32; i += 256) ((float*)sh)[i] = 0.0f;
  __syncthreads();

  const int lane = tid & 31;
  const int wave = tid >> 5;
  const int wl   = wave >> 1;
  const int dir  = wave & 1;
  const int word = word0 + wl;
  const int len  = char_lens[word];
  const float* Wih = sWih[dir];
  const float* Whh = sWhh[dir];
  const float* bb  = sb[dir];
  float* hv = sh[wl][dir];
  float cst = 0.0f;
  const bool act = (lane < CHH);

  for (int t = 0; t < L_CH; ++t) {
    const int idx = dir ? (L_CH - 1 - t) : t;     // backward runs reversed
    const bool m = (idx < len);
    const float* e = &se[wl][idx * CE];
    float h2 = 0.0f;
    if (act) {
      float gi = bb[lane], gf = bb[CHH + lane], gg = bb[2 * CHH + lane], go = bb[3 * CHH + lane];
      for (int k = 0; k < CE; ++k) {
        float ev = e[k];
        gi += Wih[lane * CE + k] * ev;
        gf += Wih[(CHH + lane) * CE + k] * ev;
        gg += Wih[(2 * CHH + lane) * CE + k] * ev;
        go += Wih[(3 * CHH + lane) * CE + k] * ev;
      }
      for (int k = 0; k < CHH; ++k) {
        float hk = hv[k];
        gi += Whh[lane * CHH + k] * hk;
        gf += Whh[(CHH + lane) * CHH + k] * hk;
        gg += Whh[(2 * CHH + lane) * CHH + k] * hk;
        go += Whh[(3 * CHH + lane) * CHH + k] * hk;
      }
      float si = sigmoidf_(gi), sf = sigmoidf_(gf);
      float sg = tanhf(gg),     so = sigmoidf_(go);
      float c2 = sf * cst + si * sg;
      h2 = so * tanhf(c2);
      if (m) cst = c2;
    }
    __syncthreads();                 // everyone done reading hv
    if (act && m) hv[lane] = h2;
    __syncthreads();
  }
  if (act) x_bf[(size_t)word * DIN_PAD + WE + dir * CHH + lane] = f2bf(hv[lane]);
}

// =====================================================================
// K4: bf16 WMMA GEMM:  Z[M x nvalid] = X[M x K](bf16) * W[N x K]^T(bf16) + bias
// One wave -> 64x32 output (4 M-tiles x 2 N-tiles). K multiple of 32.
// A/B global loads match the v_wmma_f32_16x16x32_bf16 lane layouts directly.
// =====================================================================
__global__ void __launch_bounds__(128) wmma_gemm_bf16_kernel(
    const unsigned short* __restrict__ X,
    const unsigned short* __restrict__ W,
    const float* __restrict__ bias,
    float* __restrict__ Z,
    int K, int ldz, int nvalid)
{
  const int lane  = threadIdx.x & 31;
  const int wave  = threadIdx.x >> 5;
  const int waves = blockDim.x >> 5;
  const int m0 = blockIdx.x * 64;
  const int n0 = (blockIdx.y * waves + wave) * 32;
  const int lrow = lane & 15;
  const int hi   = lane >> 4;

  const v8f vzero = {0.f, 0.f, 0.f, 0.f, 0.f, 0.f, 0.f, 0.f};
  v8f c[4][2];
#pragma unroll
  for (int i = 0; i < 4; ++i)
#pragma unroll
    for (int j = 0; j < 2; ++j) c[i][j] = vzero;

  for (int k0 = 0; k0 < K; k0 += 32) {
    V16U a[4], b[2];
    const int ak = k0 + hi * 8;        // A: lanes<16 hold K 0..7/16..23, lanes>=16 hold 8..15/24..31
#pragma unroll
    for (int i = 0; i < 4; ++i) {
      const unsigned short* pa = X + (size_t)(m0 + i * 16 + lrow) * K + ak;
      a[i].q[0] = *(const uint4*)(pa);
      a[i].q[1] = *(const uint4*)(pa + 16);
    }
    const int bk = k0 + hi * 8;        // B: lanes<16 hold K 0..7/16..23, lanes>=16 hold 8..15/24..31 (col = lane%16)
#pragma unroll
    for (int j = 0; j < 2; ++j) {
      const unsigned short* pb = W + (size_t)(n0 + j * 16 + lrow) * K + bk;
      b[j].q[0] = *(const uint4*)(pb);
      b[j].q[1] = *(const uint4*)(pb + 16);
    }
#pragma unroll
    for (int i = 0; i < 4; ++i)
#pragma unroll
      for (int j = 0; j < 2; ++j)
        c[i][j] = __builtin_amdgcn_wmma_f32_16x16x32_bf16(
            false, a[i].v, false, b[j].v, (short)0, c[i][j], false, false);
  }
  // C layout: VGPR r, lanes<16 -> M=r, lanes>=16 -> M=r+8; N = lane%16
#pragma unroll
  for (int i = 0; i < 4; ++i) {
    const int rowb = m0 + i * 16 + hi * 8;
#pragma unroll
    for (int j = 0; j < 2; ++j) {
      const int col = n0 + j * 16 + lrow;
      if (col < nvalid) {
        const float bv = bias[col];
#pragma unroll
        for (int r = 0; r < 8; ++r)
          Z[(size_t)(rowb + r) * ldz + col] = c[i][j][r] + bv;
      }
    }
  }
}

// =====================================================================
// K5: word-level LSTM recurrence (one direction), persistent 16-WG kernel.
// Each WG owns h indices [32*wg, 32*wg+32): 128 gate rows of Whh, held
// entirely in LDS (fp32, 256KB+pad) for all 4096 steps -> zero weight
// re-fetch. Per step: 2-way k-split dot, gate nonlinearity on 32 lanes,
// h exchanged through L2 with agent-scope atomics + counter barrier.
// =====================================================================
#define RWG   16
#define ROWS  128
#define WPAD  514                         // 512 + 2 -> conflict-free float2 reads
#define RSMEM ((ROWS * WPAD + HID + 256 + 128) * 4)

__global__ void __launch_bounds__(256, 1) word_lstm_kernel(
    const float* __restrict__ z,          // [S, 2048] pre-activations (x@Wih^T + b)
    const float* __restrict__ Whh,        // [2048, 512] fp32
    float* __restrict__ hbuf,             // [512] h exchange buffer
    unsigned short* __restrict__ outb,    // [S, 1024] bf16 (concat layout)
    unsigned int* __restrict__ counter,
    int S_, int rev, int outcol0)
{
  extern __shared__ float smem[];
  float* Wlds = smem;                     // ROWS * WPAD
  float* hlds = smem + ROWS * WPAD;       // 512
  float* part = hlds + HID;               // 256
  float* glds = part + 256;               // 128

  const int tid = threadIdx.x;
  const int j0  = blockIdx.x * 32;

  // one-time load of this WG's 128 gate rows (i,f,g,o chunks for h[j0..j0+32))
  for (int lr = 0; lr < ROWS; ++lr) {
    const int grow = (lr >> 5) * HID + j0 + (lr & 31);
    const float* src = Whh + (size_t)grow * HID;
    for (int k = tid; k < HID; k += 256) Wlds[lr * WPAD + k] = src[k];
  }
  for (int k = tid; k < HID; k += 256) hlds[k] = 0.0f;
  __syncthreads();

  const int half = tid >> 7;              // k-split half
  const int lr   = tid & 127;             // local gate row
  const int gidx = (lr >> 5) * HID + j0 + (lr & 31);   // global gate column in z
  const float2* wp = (const float2*)(Wlds + lr * WPAD + half * 256);
  const float2* hp = (const float2*)(hlds + half * 256);
  float cst = 0.0f;                       // cell state (valid for tid<32)

  for (int t = 0; t < S_; ++t) {
    const int s = rev ? (S_ - 1 - t) : t;
    const float zval = z[(size_t)s * G4 + gidx];

    float acc = 0.0f;
#pragma unroll 8
    for (int k = 0; k < 128; ++k) {       // 256 MACs (half of the 512-dot)
      const float2 w2 = wp[k], h2 = hp[k];
      acc += w2.x * h2.x;
      acc += w2.y * h2.y;
    }
    part[tid] = acc;
    __syncthreads();
    if (half == 0) glds[lr] = part[lr] + part[lr + 128] + zval;
    __syncthreads();

    if (tid < 32) {
      const float gi = sigmoidf_(glds[tid]);
      const float gf = sigmoidf_(glds[32 + tid]);
      const float gg = tanhf(glds[64 + tid]);
      const float go = sigmoidf_(glds[96 + tid]);
      cst = gf * cst + gi * gg;
      const float h = go * tanhf(cst);
      __hip_atomic_store(&hbuf[j0 + tid], h, __ATOMIC_RELAXED, __HIP_MEMORY_SCOPE_AGENT);
      outb[(size_t)s * D2H + outcol0 + j0 + tid] = f2bf(h);
    }
    __threadfence();                      // each thread publishes its stores
    __syncthreads();
    if (tid == 0) {
      __hip_atomic_fetch_add(counter, 1u, __ATOMIC_RELEASE, __HIP_MEMORY_SCOPE_AGENT);
      const unsigned target = (unsigned)(t + 1) * RWG;
      while (__hip_atomic_load(counter, __ATOMIC_ACQUIRE, __HIP_MEMORY_SCOPE_AGENT) < target)
        __builtin_amdgcn_s_sleep(1);
    }
    __syncthreads();
    for (int k = tid; k < HID; k += 256)  // pull full h for next step (coherent loads)
      hlds[k] = __hip_atomic_load(&hbuf[k], __ATOMIC_RELAXED, __HIP_MEMORY_SCOPE_AGENT);
    __syncthreads();
  }
}

// =====================================================================
// host-side orchestration
// =====================================================================
extern "C" void kernel_launch(void* const* d_in, const int* in_sizes, int n_in,
                              void* d_out, int out_size, void* d_ws, size_t ws_size,
                              hipStream_t stream) {
  (void)in_sizes; (void)n_in; (void)out_size;
  const int*   words    = (const int*)d_in[0];
  const int*   chars    = (const int*)d_in[1];
  const int*   char_lens= (const int*)d_in[2];
  const float* char_emb = (const float*)d_in[3];
  const float* cWih_f   = (const float*)d_in[4];
  const float* cWhh_f   = (const float*)d_in[5];
  const float* cb_f     = (const float*)d_in[6];
  const float* cWih_b   = (const float*)d_in[7];
  const float* cWhh_b   = (const float*)d_in[8];
  const float* cb_b     = (const float*)d_in[9];
  const float* word_emb = (const float*)d_in[10];
  const float* wWih_f   = (const float*)d_in[11];
  const float* wWhh_f   = (const float*)d_in[12];
  const float* wb_f     = (const float*)d_in[13];
  const float* wWih_b   = (const float*)d_in[14];
  const float* wWhh_b   = (const float*)d_in[15];
  const float* wb_b     = (const float*)d_in[16];
  const float* W_tag    = (const float*)d_in[17];
  const float* b_tag    = (const float*)d_in[18];
  float* feats = (float*)d_out;

  // ---- workspace carve-out (all offsets 256B aligned) ----
  char* ws = (char*)d_ws;
  size_t off = 0;
  auto take = [&](size_t bytes) -> char* {
    char* p = ws + off;
    off = (off + bytes + 255) & ~(size_t)255;
    return p;
  };
  unsigned short* x_bf  = (unsigned short*)take((size_t)S_LEN * DIN_PAD * 2);
  unsigned short* Wf_bf = (unsigned short*)take((size_t)G4 * DIN_PAD * 2);
  unsigned short* Wb_bf = (unsigned short*)take((size_t)G4 * DIN_PAD * 2);
  unsigned short* Wt_bf = (unsigned short*)take((size_t)NT_PAD * D2H * 2);
  float* zf             = (float*)take((size_t)S_LEN * G4 * 4);
  float* zb             = (float*)take((size_t)S_LEN * G4 * 4);
  unsigned short* out_bf= (unsigned short*)take((size_t)S_LEN * D2H * 2);
  float* hbuf_f         = (float*)take(HID * 4);
  float* hbuf_b         = (float*)take(HID * 4);
  unsigned int* counters= (unsigned int*)take(256);
  if (off > ws_size) return;   // insufficient scratch: bail deterministically

  // reset cross-call state (graph-capture-safe)
  hipMemsetAsync(counters, 0, 256, stream);

  // weight bf16 conversion (+zero padding)
  {
    int tot = G4 * DIN_PAD;
    convert_pad_kernel<<<(tot + 255) / 256, 256, 0, stream>>>(wWih_f, G4, 350, Wf_bf, G4, DIN_PAD);
    convert_pad_kernel<<<(tot + 255) / 256, 256, 0, stream>>>(wWih_b, G4, 350, Wb_bf, G4, DIN_PAD);
    int tott = NT_PAD * D2H;
    convert_pad_kernel<<<(tott + 255) / 256, 256, 0, stream>>>(W_tag, NTAG, D2H, Wt_bf, NT_PAD, D2H);
  }

  // x assembly: word embeddings + char BiLSTM reps (bf16)
  gather_x_kernel<<<S_LEN, 128, 0, stream>>>(words, word_emb, x_bf);
  char_bilstm_kernel<<<S_LEN / 4, 256, 0, stream>>>(
      chars, char_lens, char_emb, cWih_f, cWhh_f, cb_f, cWih_b, cWhh_b, cb_b, x_bf);

  // input projections z = x @ Wih^T + b  (bf16 WMMA, fp32 out)
  {
    dim3 g(S_LEN / 64, G4 / (32 * 4));
    wmma_gemm_bf16_kernel<<<g, 128, 0, stream>>>(x_bf, Wf_bf, wb_f, zf, DIN_PAD, G4, G4);
    wmma_gemm_bf16_kernel<<<g, 128, 0, stream>>>(x_bf, Wb_bf, wb_b, zb, DIN_PAD, G4, G4);
  }

  // word-level recurrence: LDS-resident Whh, 16 co-resident WGs per direction
  hipFuncSetAttribute(reinterpret_cast<const void*>(word_lstm_kernel),
                      hipFuncAttributeMaxDynamicSharedMemorySize, (int)RSMEM);
  word_lstm_kernel<<<RWG, 256, RSMEM, stream>>>(zf, wWhh_f, hbuf_f, out_bf, counters + 0,  S_LEN, 0, 0);
  word_lstm_kernel<<<RWG, 256, RSMEM, stream>>>(zb, wWhh_b, hbuf_b, out_bf, counters + 32, S_LEN, 1, HID);

  // tag head: feats = out_bf @ W_tag^T + b_tag  (bf16 WMMA, fp32 out)
  {
    dim3 g(S_LEN / 64, 1);
    wmma_gemm_bf16_kernel<<<g, 64, 0, stream>>>(out_bf, Wt_bf, b_tag, feats, D2H, NTAG, NTAG);
  }
}
